// SSMNet_74139725463688
// MI455X (gfx1250) — matrix operand order repaired and checked by reading, not compiled, not run on hardware
//
#include <hip/hip_runtime.h>

// Problem constants (from reference)
#define B_ 16
#define T_ 1024
#define I_ 256
#define H_ 512
#define L_ 4
#define S_ 64

typedef __attribute__((ext_vector_type(16))) __bf16 v16bf;
typedef __attribute__((ext_vector_type(8)))  float  v8f;

union Frag16 { unsigned int u[8]; v16bf v; };

__device__ __forceinline__ unsigned short f2bf(float f) {
    unsigned int u = __float_as_uint(f);
    unsigned int r = u + 0x7FFFu + ((u >> 16) & 1u);   // round-to-nearest-even
    return (unsigned short)(r >> 16);
}

// ---------------------------------------------------------------------------
// One-shot f32 -> packed bf16 conversion (8 elements / thread).
// ---------------------------------------------------------------------------
__global__ __launch_bounds__(256) void cvt_bf16_kernel(
    const float* __restrict__ in, unsigned short* __restrict__ out, int n8)
{
    int i = blockIdx.x * 256 + threadIdx.x;
    if (i >= n8) return;
    const float4* p = (const float4*)in + (size_t)i * 2;
    float4 a = p[0], b = p[1];
    union { unsigned int u[4]; uint4 v; } r;
    r.u[0] = ((unsigned)f2bf(a.y) << 16) | f2bf(a.x);
    r.u[1] = ((unsigned)f2bf(a.w) << 16) | f2bf(a.z);
    r.u[2] = ((unsigned)f2bf(b.y) << 16) | f2bf(b.x);
    r.u[3] = ((unsigned)f2bf(b.w) << 16) | f2bf(b.z);
    ((uint4*)out)[i] = r.v;
}

// ---------------------------------------------------------------------------
// FC GEMM: out[b*H+h][t] = sum_i x[b*T+t, i] * w[h, i] + bias[h]
// M = B*T = 16384, N = H = 512, K = I = 256, bf16 inputs pre-converted.
// Block tile 128x64, 8 waves (4 M x 2 N), each wave 32x32 via four
// v_wmma_f32_16x16x32_bf16 accumulators. Tiles staged to LDS with
// GLOBAL_LOAD_ASYNC_TO_LDS_B128 when available (ASYNCcnt path).
// Rows padded to 80 B: 16B-aligned for B128 async writes AND bank-clean
// (20 dwords/row -> 16 fragment rows hit 16 distinct banks).
// ---------------------------------------------------------------------------
#define BM 128
#define BN 64
#define BK 32
#define LDA 40
#define LDB 40

#if __has_builtin(__builtin_amdgcn_global_load_async_to_lds_b128)
#define HAS_ASYNC_LDS 1
// Toolchain signature (from diagnostics):
//   arg0 = int4* addrspace(1), arg1 = int4* addrspace(3), arg2 = imm, arg3 = cpol
typedef int v4i_ __attribute__((vector_size(16)));
typedef __attribute__((address_space(1))) v4i_ g_v4i;
typedef __attribute__((address_space(3))) v4i_ l_v4i;
#else
#define HAS_ASYNC_LDS 0
#endif

__global__ __launch_bounds__(256) void fc_wmma_kernel(
    const unsigned short* __restrict__ xb,   // (B*T, I) bf16
    const unsigned short* __restrict__ wb,   // (H, I)  bf16
    const float* __restrict__ bias,          // (H)
    float* __restrict__ out)                 // (B*H, T)
{
    __shared__ unsigned short As[BM * LDA];
    __shared__ unsigned short Bs[BN * LDB];

    const int tid    = threadIdx.x;
    const int lane   = tid & 31;
    const int wave   = tid >> 5;        // 0..7
    const int wm     = wave & 3;        // 4 waves along M
    const int wn     = wave >> 2;       // 2 waves along N
    const int blockM = blockIdx.y * BM;
    const int blockN = blockIdx.x * BN;
    const int half   = lane >> 4;
    const int r16    = lane & 15;

    v8f acc[2][2] = {};

    for (int k0 = 0; k0 < I_; k0 += BK) {
#if HAS_ASYNC_LDS
        // A tile: 128 rows x 64 B = 512 x 16B chunks; 2 per thread.
        #pragma unroll
        for (int cc = 0; cc < 2; ++cc) {
            int c   = tid + cc * 256;
            int row = c >> 2, j = c & 3;
            __builtin_amdgcn_global_load_async_to_lds_b128(
                (g_v4i*)(xb + (size_t)(blockM + row) * I_ + k0 + j * 8),
                (l_v4i*)&As[row * LDA + j * 8], 0, 0);
        }
        // B tile: 64 rows x 64 B = 256 x 16B chunks; 1 per thread.
        {
            int row = tid >> 2, j = tid & 3;
            __builtin_amdgcn_global_load_async_to_lds_b128(
                (g_v4i*)(wb + (size_t)(blockN + row) * I_ + k0 + j * 8),
                (l_v4i*)&Bs[row * LDB + j * 8], 0, 0);
        }
#if __has_builtin(__builtin_amdgcn_s_wait_asynccnt)
        __builtin_amdgcn_s_wait_asynccnt(0);
#else
        asm volatile("s_wait_asynccnt 0x0" ::: "memory");
#endif
#else
        // Fallback: plain 16 B vector copies via VGPRs.
        #pragma unroll
        for (int cc = 0; cc < 2; ++cc) {
            int c   = tid + cc * 256;
            int row = c >> 2, j = c & 3;
            *(uint4*)&As[row * LDA + j * 8] =
                *(const uint4*)(xb + (size_t)(blockM + row) * I_ + k0 + j * 8);
        }
        {
            int row = tid >> 2, j = tid & 3;
            *(uint4*)&Bs[row * LDB + j * 8] =
                *(const uint4*)(wb + (size_t)(blockN + row) * I_ + k0 + j * 8);
        }
#endif
        __syncthreads();

        // A fragments: 16-bit 16x32 layout — lane half splits K {0..7,16..23}/{8..15,24..31}
        Frag16 fa[2], fb[2];
        #pragma unroll
        for (int mt = 0; mt < 2; ++mt) {
            int row = wm * 32 + mt * 16 + r16;
            #pragma unroll
            for (int i = 0; i < 8; ++i) {
                int k = (i < 4) ? (half * 8 + 2 * i) : (16 + half * 8 + 2 * (i - 4));
                fa[mt].u[i] = *(const unsigned int*)&As[row * LDA + k];
            }
        }
        // B fragments: lane = column n, K packed 2/dword, halves split K {0..7,16..23}/{8..15,24..31}
        #pragma unroll
        for (int nt = 0; nt < 2; ++nt) {
            int col = wn * 32 + nt * 16 + r16;
            #pragma unroll
            for (int i = 0; i < 8; ++i) {
                int k = (i < 4) ? (half * 8 + 2 * i) : (16 + half * 8 + 2 * (i - 4));
                fb[nt].u[i] = *(const unsigned int*)&Bs[col * LDB + k];
            }
        }

        #pragma unroll
        for (int mt = 0; mt < 2; ++mt)
            #pragma unroll
            for (int nt = 0; nt < 2; ++nt)
                acc[mt][nt] = __builtin_amdgcn_wmma_f32_16x16x32_bf16(
                    false, fa[mt].v, false, fb[nt].v,
                    (short)0, acc[mt][nt], false, false);

        __syncthreads();
    }

    // Epilogue: D layout — vgpr r: M = r + 8*half, N = lane&15. Store to (B*H, T).
    #pragma unroll
    for (int nt = 0; nt < 2; ++nt) {
        int n    = blockN + wn * 32 + nt * 16 + r16;   // h
        float bv = bias[n];
        #pragma unroll
        for (int mt = 0; mt < 2; ++mt) {
            #pragma unroll
            for (int r = 0; r < 8; ++r) {
                int m  = blockM + wm * 32 + mt * 16 + half * 8 + r;  // b*T + t
                int bb = m >> 10;        // / T_
                int tt = m & (T_ - 1);
                out[((size_t)bb * H_ + n) * T_ + tt] = acc[mt][nt][r] + bv;
            }
        }
    }
}

// ---------------------------------------------------------------------------
// S4D diagonal scan + spike. One wave32 per (b,h); 2 complex states per lane.
// Reads u in (B*H, T) layout: one contiguous 32-float chunk per 32 steps,
// broadcast per step with __shfl; butterfly __shfl_xor reduction over S=64.
// Writes (B*H, T) for intermediate layers, (B, T, H) for the final layer.
// ---------------------------------------------------------------------------
__global__ __launch_bounds__(256) void s4d_scan_kernel(
    const float* __restrict__ u_in,
    float* __restrict__ u_out,
    const float* __restrict__ log_dt, const float* __restrict__ logA_re,
    const float* __restrict__ A_imag, const float* __restrict__ C_re,
    const float* __restrict__ C_im,   const float* __restrict__ Dp,
    int layer, int last)
{
    const int lane = threadIdx.x & 31;
    const int bh   = blockIdx.x * 8 + (threadIdx.x >> 5);
    const int b    = bh / H_;
    const int h    = bh - b * H_;

    const float dt = expf(log_dt[layer * H_ + h]);

    float dAr[2], dAi[2], dBr[2], dBi[2], Cr[2], Ci[2];
    #pragma unroll
    for (int q = 0; q < 2; ++q) {
        int s = 2 * lane + q;
        size_t idx = ((size_t)layer * H_ + h) * S_ + s;
        float Ar = -expf(logA_re[idx]);
        float Ai = A_imag[idx];
        float er = expf(dt * Ar);
        float sn, cs;
        __sincosf(dt * Ai, &sn, &cs);
        dAr[q] = er * cs;
        dAi[q] = er * sn;
        float nr  = dAr[q] - 1.0f, ni = dAi[q];
        float inv = 1.0f / (Ar * Ar + Ai * Ai);
        dBr[q] = (nr * Ar + ni * Ai) * inv;   // (dA-1)/A via conj(A)/|A|^2
        dBi[q] = (ni * Ar - nr * Ai) * inv;
        Cr[q]  = C_re[idx];
        Ci[q]  = C_im[idx];
    }
    const float Dv = Dp[layer * H_ + h];

    float zr[2] = {0.0f, 0.0f}, zi[2] = {0.0f, 0.0f};
    const float* uptr = u_in + (size_t)bh * T_;

    for (int t0 = 0; t0 < T_; t0 += 32) {
        float uch = uptr[t0 + lane];
        if (t0 + 32 < T_)
            __builtin_prefetch(uptr + t0 + 32 + lane, 0, 0);  // global_prefetch_b8
        float outv = 0.0f;
        #pragma unroll 8
        for (int j = 0; j < 32; ++j) {
            float u = __shfl(uch, j, 32);
            #pragma unroll
            for (int q = 0; q < 2; ++q) {
                float nzr = dAr[q] * zr[q] - dAi[q] * zi[q] + dBr[q] * u;
                float nzi = dAr[q] * zi[q] + dAi[q] * zr[q] + dBi[q] * u;
                zr[q] = nzr; zi[q] = nzi;
            }
            float part = zr[0] * Cr[0] - zi[0] * Ci[0]
                       + zr[1] * Cr[1] - zi[1] * Ci[1];
            #pragma unroll
            for (int off = 16; off > 0; off >>= 1)
                part += __shfl_xor(part, off, 32);
            float y  = 2.0f * part + Dv * u;
            float sp = (y > 0.0f) ? 1.0f : 0.0f;
            outv = (j == lane) ? sp : outv;      // lane j keeps step t0+j
        }
        if (last) {
            int t = t0 + lane;
            u_out[((size_t)b * T_ + t) * H_ + h] = outv;   // (B, T, H)
        } else {
            u_out[(size_t)bh * T_ + t0 + lane] = outv;     // (B*H, T)
        }
    }
}

// ---------------------------------------------------------------------------
extern "C" void kernel_launch(void* const* d_in, const int* in_sizes, int n_in,
                              void* d_out, int out_size, void* d_ws, size_t ws_size,
                              hipStream_t stream) {
    (void)in_sizes; (void)n_in; (void)out_size; (void)ws_size;

    const float* x      = (const float*)d_in[0];
    const float* fc_w   = (const float*)d_in[1];
    const float* fc_b   = (const float*)d_in[2];
    const float* log_dt = (const float*)d_in[3];
    const float* logA   = (const float*)d_in[4];
    const float* A_im   = (const float*)d_in[5];
    const float* C_re   = (const float*)d_in[6];
    const float* C_im   = (const float*)d_in[7];
    const float* Dp     = (const float*)d_in[8];

    const size_t BHT = (size_t)B_ * H_ * T_;
    float* u0 = (float*)d_ws;                         // 32 MB
    float* u1 = u0 + BHT;                             // 32 MB
    unsigned short* xb = (unsigned short*)(u1 + BHT); // 8 MB bf16 x
    unsigned short* wb = xb + (size_t)B_ * T_ * I_;   // 256 KB bf16 w

    // Pre-convert inputs to bf16 (once; tiles then re-read from L2).
    {
        int n8x = (B_ * T_ * I_) / 8;
        int n8w = (H_ * I_) / 8;
        cvt_bf16_kernel<<<dim3((n8x + 255) / 256), 256, 0, stream>>>(x, xb, n8x);
        cvt_bf16_kernel<<<dim3((n8w + 255) / 256), 256, 0, stream>>>(fc_w, wb, n8w);
    }

    dim3 ggrid(H_ / BN, (B_ * T_) / BM);              // (8, 128)
    fc_wmma_kernel<<<ggrid, 256, 0, stream>>>(xb, wb, fc_b, u0);

    const float* in = u0;
    float* pong[2] = { u1, u0 };
    for (int l = 0; l < L_; ++l) {
        int last = (l == L_ - 1);
        float* outp = last ? (float*)d_out : pong[l & 1];
        s4d_scan_kernel<<<dim3(B_ * H_ / 8), 256, 0, stream>>>(
            in, outp, log_dt, logA, A_im, C_re, C_im, Dp, l, last);
        in = outp;
    }
}
